// MultiQueryAttentionBlock_66992899883044
// MI455X (gfx1250) — matrix operand order repaired and checked by reading, not compiled, not run on hardware
//
#include <hip/hip_runtime.h>
#include <hip/hip_bf16.h>

typedef __attribute__((ext_vector_type(16))) _Float16 v16h;
typedef __attribute__((ext_vector_type(8)))  float    v8f;
typedef __attribute__((ext_vector_type(4)))  int      i4v;

#if defined(__has_builtin)
#if __has_builtin(__builtin_amdgcn_global_load_async_to_lds_b128)
#define HAVE_ASYNC_LDS 1
#endif
#endif
#ifndef HAVE_ASYNC_LDS
#define HAVE_ASYNC_LDS 0
#endif

// ---------------------------------------------------------------------------
// Async global->LDS copies (gfx1250 GLOBAL_LOAD_ASYNC_TO_LDS_B*, ASYNCcnt)
// Builtin prototype (from hipcc diagnostics): b128 takes int4* args, b32 int*,
// non-const. Order follows llvm.amdgcn.global.load.lds: (global, lds, off, cpol).
// ---------------------------------------------------------------------------
__device__ __forceinline__ void async_b128(const float* g, float* l) {
#if HAVE_ASYNC_LDS
  __builtin_amdgcn_global_load_async_to_lds_b128(
      (__attribute__((address_space(1))) i4v*)(i4v*)g,
      (__attribute__((address_space(3))) i4v*)(i4v*)l, 0, 0);
#else
  *(float4*)l = *(const float4*)g;
#endif
}
__device__ __forceinline__ void async_b32(const float* g, float* l) {
#if HAVE_ASYNC_LDS
  __builtin_amdgcn_global_load_async_to_lds_b32(
      (__attribute__((address_space(1))) int*)(int*)g,
      (__attribute__((address_space(3))) int*)(int*)l, 0, 0);
#else
  *l = *g;
#endif
}
__device__ __forceinline__ void wait_async0() {
#if HAVE_ASYNC_LDS
  asm volatile("s_wait_asynccnt 0" ::: "memory");
#endif
}

// ---------------------------------------------------------------------------
// WMMA helpers (gfx1250: D(f32 16x16) = A(f16 16x32) x B(f16 32x16) + C)
// ---------------------------------------------------------------------------
__device__ __forceinline__ v8f wmma_f16(v16h a, v16h b, v8f c) {
  return __builtin_amdgcn_wmma_f32_16x16x32_f16(
      false, a, false, b, (short)0, c, false, false);
}

// A-fragment from 32 contiguous f16 K values for row (lane&15).
__device__ __forceinline__ v16h a_frag_h(const _Float16* row, int hi) {
  v16h a;
#pragma unroll
  for (int i = 0; i < 8; ++i) {
    a[i]     = row[hi * 16 + i];
    a[8 + i] = row[hi * 16 + 8 + i];
  }
  return a;
}
// A-fragment from 32 contiguous f32 K values (convert on the fly).
__device__ __forceinline__ v16h a_frag_f(const float* row, int hi) {
  v16h a;
#pragma unroll
  for (int i = 0; i < 8; ++i) {
    a[i]     = (_Float16)row[hi * 16 + i];
    a[8 + i] = (_Float16)row[hi * 16 + 8 + i];
  }
  return a;
}
// B-fragment from 16 contiguous f32 (transposed [N][K] view), global (float4).
__device__ __forceinline__ v16h b_frag_g(const float* src) {
  v16h b;
  const float4* p = (const float4*)src;
#pragma unroll
  for (int q = 0; q < 4; ++q) {
    float4 v = p[q];
    b[4 * q + 0] = (_Float16)v.x;
    b[4 * q + 1] = (_Float16)v.y;
    b[4 * q + 2] = (_Float16)v.z;
    b[4 * q + 3] = (_Float16)v.w;
  }
  return b;
}
// B-fragment from 16 contiguous f32 in LDS.
__device__ __forceinline__ v16h b_frag_l(const float* p) {
  v16h b;
#pragma unroll
  for (int i = 0; i < 16; ++i) b[i] = (_Float16)p[i];
  return b;
}

// ---------------------------------------------------------------------------
// Tiled WMMA GEMM: C[M,N] = A[M,K] @ B[K,N] + bias (f32 memory, f16 compute).
// Block tile 128x32, 8 waves, k-step 32; LDS staged via async-to-LDS copies.
// flags: 1 = store C transposed (C[n*M+m]); 2 = non-temporal stores.
// Requires M%128==0, N%32==0, K%32==0.
// ---------------------------------------------------------------------------
__global__ __launch_bounds__(256) void gemm_wmma(
    const float* __restrict__ A, const float* __restrict__ B,
    const float* __restrict__ bias, float* __restrict__ C,
    int M, int N, int K, int flags) {
  __shared__ float Alf[128 * 36];  // [128 rows][32 k + pad]  (f32)
  __shared__ float Btf[32 * 36];   // transposed [32 n][32 k + pad]

  const int tid  = threadIdx.x;
  const int wave = tid >> 5;
  const int lane = tid & 31;
  const int l15  = lane & 15;
  const int hi   = lane >> 4;
  const int mbase = blockIdx.x * 128;
  const int nbase = blockIdx.y * 32;

  v8f acc0 = {};
  v8f acc1 = {};

  for (int kb = 0; kb < K; kb += 32) {
    __syncthreads();  // previous iteration's fragment reads done
    // Stage A tile 128x32: 4x B128 async per thread, contiguous rows.
#pragma unroll
    for (int i = 0; i < 4; ++i) {
      int idx = tid + i * 256;  // float4 index
      int r = idx >> 3, c = (idx & 7) * 4;
      async_b128(A + (size_t)(mbase + r) * K + kb + c, &Alf[r * 36 + c]);
    }
    // Stage B tile 32x32 transposed: 4x B32 async per thread (coalesced reads).
#pragma unroll
    for (int j = 0; j < 4; ++j) {
      int e = tid + j * 256;  // element index
      int k = e >> 5, c = e & 31;
      async_b32(B + (size_t)(kb + k) * N + nbase + c, &Btf[c * 36 + k]);
    }
    wait_async0();
    __syncthreads();

    v16h a  = a_frag_f(&Alf[(wave * 16 + l15) * 36], hi);
    v16h b0 = b_frag_l(&Btf[l15 * 36 + hi * 16]);
    v16h b1 = b_frag_l(&Btf[(16 + l15) * 36 + hi * 16]);
    acc0 = wmma_f16(a, b0, acc0);
    acc1 = wmma_f16(a, b1, acc1);
  }

  const float bia0 = bias[nbase + l15];
  const float bia1 = bias[nbase + 16 + l15];
  if (flags & 1) {  // transposed store (Vt)
#pragma unroll
    for (int v = 0; v < 8; ++v) {
      int row = mbase + wave * 16 + v + 8 * hi;
      C[(size_t)(nbase + l15) * M + row]      = acc0[v] + bia0;
      C[(size_t)(nbase + 16 + l15) * M + row] = acc1[v] + bia1;
    }
  } else if (flags & 2) {  // streaming output, never re-read: NT stores
#pragma unroll
    for (int v = 0; v < 8; ++v) {
      size_t row = (size_t)(mbase + wave * 16 + v + 8 * hi) * N;
      __builtin_nontemporal_store(acc0[v] + bia0, &C[row + nbase + l15]);
      __builtin_nontemporal_store(acc1[v] + bia1, &C[row + nbase + 16 + l15]);
    }
  } else {
#pragma unroll
    for (int v = 0; v < 8; ++v) {
      size_t row = (size_t)(mbase + wave * 16 + v + 8 * hi) * N;
      C[row + nbase + l15]      = acc0[v] + bia0;
      C[row + nbase + 16 + l15] = acc1[v] + bia1;
    }
  }
}

// ---------------------------------------------------------------------------
// Fused multi-query attention core for one (b, h, 16-query-row) tile.
//   pass 1: energy via WMMA, online row max / exp-sum (no energy storage)
//   pass 2: recompute energy, normalize, stream f32 attention to HBM once
//           (non-temporal), accumulate x = P @ V via WMMA (V transposed).
// Q: [4096,1024] (b*2048+l, h*64+d)   Kp: [4096,64]   Vt: [64,4096]
// attn: [B*H, 2048, 2048]             X: [4096,1024]
// ---------------------------------------------------------------------------
__global__ __launch_bounds__(256) void mqa_attn(
    const float* __restrict__ Q, const float* __restrict__ Kp,
    const float* __restrict__ Vt, float* __restrict__ attn,
    float* __restrict__ X) {
  __shared__ _Float16 Qh[16 * 72];       // Q tile 16x64 f16 (+pad)
  __shared__ _Float16 Pst[8][16 * 40];   // per-wave P staging 16x32 f16 (+pad)
  __shared__ float    RedM[8][16];
  __shared__ float    RedS[8][16];
  __shared__ float    Xp[8][16 * 64];    // per-wave partial x

  const int tid  = threadIdx.x;
  const int wave = tid >> 5;
  const int lane = tid & 31;
  const int l15  = lane & 15;
  const int hi   = lane >> 4;

  const int qt = blockIdx.x;  // 0..127
  const int bh = blockIdx.y;  // 0..31
  const int b  = bh >> 4;
  const int h  = bh & 15;
  const int qbase = qt * 16;
  const size_t qrow0 = (size_t)(b * 2048 + qbase);

  // Stage Q 16x64 -> f16 in LDS
  {
    int r = tid >> 4, c = (tid & 15) * 4;
    float4 v = *(const float4*)(Q + (qrow0 + r) * 1024 + h * 64 + c);
    _Float16* d = Qh + r * 72 + c;
    d[0] = (_Float16)v.x; d[1] = (_Float16)v.y;
    d[2] = (_Float16)v.z; d[3] = (_Float16)v.w;
  }
  __syncthreads();

  v16h qa0 = a_frag_h(Qh + l15 * 72, hi);       // d = 0..31
  v16h qa1 = a_frag_h(Qh + l15 * 72 + 32, hi);  // d = 32..63

  const float* Kbase = Kp + (size_t)(b * 2048) * 64;
  const float* VtB   = Vt + (size_t)(b * 2048);
  const int jw = wave * 256;  // this wave's 256-key slice

  float mrow[8], srow[8];
#pragma unroll
  for (int v = 0; v < 8; ++v) { mrow[v] = -3.0e38f; srow[v] = 0.0f; }

  // ---- pass 1: online max / exp-sum over this wave's 16 key tiles ----
  for (int jt = 0; jt < 16; ++jt) {
    int jb = jw + jt * 16;
    const float* krow = Kbase + (size_t)(jb + l15) * 64 + hi * 16;
    if (jt + 1 < 16)  // global_prefetch_b8 for the next K tile
      __builtin_prefetch(Kbase + (size_t)(jb + 16 + l15) * 64 + hi * 16, 0, 3);
    v8f e = {};
    e = wmma_f16(qa0, b_frag_g(krow), e);
    e = wmma_f16(qa1, b_frag_g(krow + 32), e);
#pragma unroll
    for (int v = 0; v < 8; ++v) {
      float ev = e[v] * 0.125f;  // 1/sqrt(64)
      float mn = fmaxf(mrow[v], ev);
      srow[v] = srow[v] * __expf(mrow[v] - mn) + __expf(ev - mn);
      mrow[v] = mn;
    }
  }

  // Reduce (m,s) across the 16 lanes holding each row (xor <=8 stays in half)
#pragma unroll
  for (int mask = 1; mask <= 8; mask <<= 1) {
#pragma unroll
    for (int v = 0; v < 8; ++v) {
      float m2 = __shfl_xor(mrow[v], mask, 32);
      float s2 = __shfl_xor(srow[v], mask, 32);
      float mn = fmaxf(mrow[v], m2);
      srow[v] = srow[v] * __expf(mrow[v] - mn) + s2 * __expf(m2 - mn);
      mrow[v] = mn;
    }
  }
  if (l15 == 0) {
#pragma unroll
    for (int v = 0; v < 8; ++v) {
      RedM[wave][v + 8 * hi] = mrow[v];
      RedS[wave][v + 8 * hi] = srow[v];
    }
  }
  __syncthreads();

  float Mf[8], Sinv[8];
#pragma unroll
  for (int v = 0; v < 8; ++v) {
    int r = v + 8 * hi;
    float M = RedM[0][r], S = RedS[0][r];
#pragma unroll
    for (int w = 1; w < 8; ++w) {
      float m2 = RedM[w][r], s2 = RedS[w][r];
      float mn = fmaxf(M, m2);
      S = S * __expf(M - mn) + s2 * __expf(m2 - mn);
      M = mn;
    }
    Mf[v] = M;
    Sinv[v] = 1.0f / S;
  }

  // ---- pass 2: recompute energy, stream attention out (NT), accumulate x ---
  v8f x0 = {}, x1 = {}, x2 = {}, x3 = {};
  float* attnBase = attn + ((size_t)bh * 2048 + qbase) * 2048;

  for (int jt2 = 0; jt2 < 8; ++jt2) {
#pragma unroll
    for (int sub = 0; sub < 2; ++sub) {
      int jb = jw + (jt2 * 2 + sub) * 16;
      const float* krow = Kbase + (size_t)(jb + l15) * 64 + hi * 16;
      v8f e = {};
      e = wmma_f16(qa0, b_frag_g(krow), e);
      e = wmma_f16(qa1, b_frag_g(krow + 32), e);
#pragma unroll
      for (int v = 0; v < 8; ++v) {
        float p = __expf(e[v] * 0.125f - Mf[v]) * Sinv[v];
        // 537 MB stream, never re-read on device -> non-temporal store
        __builtin_nontemporal_store(
            p, &attnBase[(size_t)(v + 8 * hi) * 2048 + jb + l15]);
        Pst[wave][(v + 8 * hi) * 40 + sub * 16 + l15] = (_Float16)p;
      }
    }
    // Per-wave LDS RAW: LDS ops are in-order per wave; fence compiler + DS.
    asm volatile("s_wait_dscnt 0" ::: "memory");
    __builtin_amdgcn_wave_barrier();

    v16h pa = a_frag_h(&Pst[wave][l15 * 40], hi);  // P 16x32 (j = k-dim)
    int jj = jw + jt2 * 32;
    const float* vt0 = VtB + (size_t)l15 * 4096 + jj + hi * 16;
    x0 = wmma_f16(pa, b_frag_g(vt0),             x0);
    x1 = wmma_f16(pa, b_frag_g(vt0 + 16 * 4096), x1);
    x2 = wmma_f16(pa, b_frag_g(vt0 + 32 * 4096), x2);
    x3 = wmma_f16(pa, b_frag_g(vt0 + 48 * 4096), x3);
    __builtin_amdgcn_wave_barrier();
  }

  // Cross-wave reduction of partial x (each wave covered 256 keys)
#pragma unroll
  for (int v = 0; v < 8; ++v) {
    int r = v + 8 * hi;
    Xp[wave][r * 64 +      l15] = x0[v];
    Xp[wave][r * 64 + 16 + l15] = x1[v];
    Xp[wave][r * 64 + 32 + l15] = x2[v];
    Xp[wave][r * 64 + 48 + l15] = x3[v];
  }
  __syncthreads();
#pragma unroll
  for (int i = 0; i < 4; ++i) {
    int idx = tid + i * 256;
    int r = idx >> 6, d = idx & 63;
    float s = 0.0f;
#pragma unroll
    for (int w = 0; w < 8; ++w) s += Xp[w][r * 64 + d];
    X[(qrow0 + r) * 1024 + h * 64 + d] = s;
  }
}

// ---------------------------------------------------------------------------
// Launcher: 5 stream-ordered kernels
// ---------------------------------------------------------------------------
extern "C" void kernel_launch(void* const* d_in, const int* in_sizes, int n_in,
                              void* d_out, int out_size, void* d_ws,
                              size_t ws_size, hipStream_t stream) {
  (void)in_sizes; (void)n_in; (void)out_size; (void)ws_size;
  const float* query = (const float*)d_in[0];
  const float* key   = (const float*)d_in[1];
  const float* value = (const float*)d_in[2];
  const float* Wq    = (const float*)d_in[3];
  const float* bq    = (const float*)d_in[4];
  const float* Wk    = (const float*)d_in[5];
  const float* bk    = (const float*)d_in[6];
  const float* Wv    = (const float*)d_in[7];
  const float* bv    = (const float*)d_in[8];
  const float* Wo    = (const float*)d_in[9];
  const float* bo    = (const float*)d_in[10];

  float* out  = (float*)d_out;                  // [2,2048,1024]
  float* attn = out + (size_t)2 * 2048 * 1024;  // [2,16,2048,2048]

  float* Qws = (float*)d_ws;                    // 4096 x 1024
  float* Kws = Qws + (size_t)4096 * 1024;       // 4096 x 64
  float* Vtw = Kws + (size_t)4096 * 64;         // 64 x 4096 (transposed)
  float* Xws = Vtw + (size_t)4096 * 64;         // 4096 x 1024

  dim3 blk(256);
  gemm_wmma<<<dim3(32, 32), blk, 0, stream>>>(query, Wq, bq, Qws, 4096, 1024, 1024, 0);
  gemm_wmma<<<dim3(32, 2),  blk, 0, stream>>>(key,   Wk, bk, Kws, 4096,   64, 1024, 0);
  gemm_wmma<<<dim3(32, 2),  blk, 0, stream>>>(value, Wv, bv, Vtw, 4096,   64, 1024, 1);
  mqa_attn<<<dim3(128, 32), blk, 0, stream>>>(Qws, Kws, Vtw, attn, Xws);
  gemm_wmma<<<dim3(32, 32), blk, 0, stream>>>(Xws, Wo, bo, out, 4096, 1024, 1024, 2);
}
